// SPGCL_21844203668132
// MI455X (gfx1250) — compile-verified, matrix operand order checked by reading, not run on hardware
//
#include <hip/hip_runtime.h>
#include <hip/hip_bf16.h>

// ---------------------------------------------------------------------------
// GAT layer for MI455X (gfx1250, wave32).
// Inputs (dict order): x[N,128] f32, edge_index[2,E] int, W_proj[128,128] f32,
// score_src[1,4,32] f32, score_tgt[1,4,32] f32, W_skip[128,128] f32, bias[128] f32
// Output: elu(segment_sum(att * proj[src]) + x@W_skip + bias)  -> [N,128] f32
// ---------------------------------------------------------------------------

typedef __attribute__((ext_vector_type(16))) __bf16          bf16x16;
typedef __attribute__((ext_vector_type(8)))  float           v8f;
typedef __attribute__((ext_vector_type(4)))  unsigned int    u32x4;
typedef __attribute__((ext_vector_type(8)))  int             i32x8;
typedef __attribute__((ext_vector_type(4)))  int             i32x4;

#define F_TOT 128   // H*F == F_IN == 128
#define NH    4

static __device__ __forceinline__ unsigned short f2bf(float f) {
    unsigned u = __builtin_bit_cast(unsigned, f);
    u += 0x7FFFu + ((u >> 16) & 1u);   // round-to-nearest-even
    return (unsigned short)(u >> 16);
}

union BFrag {
    unsigned short s[16];
    uint4          q[2];
    bf16x16        v;
};

// ---------------------------------------------------------------------------
// Kernel 1: convert W_proj / W_skip (f32, row-major [K=128][C=128]) into
// bf16 column-major [C][K] so B fragments are two contiguous 16B loads.
// ---------------------------------------------------------------------------
__global__ void prep_weights(const float* __restrict__ Wp, const float* __restrict__ Ws,
                             unsigned short* __restrict__ WpT, unsigned short* __restrict__ WsT) {
    int t = blockIdx.x * blockDim.x + threadIdx.x;
    if (t >= 128 * 128) return;
    int k = t >> 7, c = t & 127;
    WpT[c * 128 + k] = f2bf(Wp[t]);
    WsT[c * 128 + k] = f2bf(Ws[t]);
}

// ---------------------------------------------------------------------------
// TDM helper: issue a 2-D tensor_load_to_lds of a 128x128 bf16 tensor into
// LDS.  D# layout per CDNA5 ISA 8.3/8.4: group0 = {flags, lds_addr,
// global_addr[31:0], global_addr[56:32] | type=2}, group1 packs data_size,
// tensor_dim0/1, tile_dim0/1/2, tensor_dim0_stride.  This toolchain's builtin
// takes 6 args: (g0, g1, g2, g3, g4, cpol).
// ---------------------------------------------------------------------------
static __device__ __forceinline__ void tdm_load_128x128_bf16(unsigned lds_off,
                                                             const void* gptr) {
    unsigned long long ga = (unsigned long long)gptr;
    u32x4 g0 = { 1u,                                   // count=1, user desc
                 lds_off,                              // LDS byte address
                 (unsigned)ga,                         // global addr [31:0]
                 (unsigned)((ga >> 32) & 0x01FFFFFFu)  // global addr [56:32]
                   | 0x80000000u };                    // type=2 ("image")
    i32x8 g1 = { (int)(1u << 16),       // data_size=1 -> 2-byte elements
                 (int)(128u << 16),     // tensor_dim0[15:0] = 128
                 (int)(128u << 16),     // tensor_dim0 hi=0 | tensor_dim1 lo=128
                 (int)(128u << 16),     // tensor_dim1 hi=0 | tile_dim0=128
                 128,                   // tile_dim1=128, tile_dim2=0
                 128,                   // tensor_dim0_stride[31:0] = 128
                 0, 0 };                // stride0 hi, tensor_dim1_stride (unused)
    i32x4 g2 = { 0, 0, 0, 0 };
    i32x4 g3 = { 0, 0, 0, 0 };
    i32x8 g4 = { 0, 0, 0, 0, 0, 0, 0, 0 };
    __builtin_amdgcn_tensor_load_to_lds(g0, g1, g2, g3, g4, 0);
}

// ---------------------------------------------------------------------------
// Kernel 2: proj = x @ W_proj ; acc = x @ W_skip   (WMMA bf16 -> f32)
// W tiles staged into LDS once per workgroup via the Tensor Data Mover;
// one wave owns a 16-row stripe, all 128 output columns, both matrices.
// A-fragment (16-bit A 16x32, ISA 7.12.2): lane<16 row=lane, k = {0..7,16..23}
// (+8 for lanes 16..31).  C/D: VGPR j -> row = j + 8*(lane>=16), col = lane&15.
// ---------------------------------------------------------------------------
__global__ void __launch_bounds__(256)
gemm_proj_skip(const float* __restrict__ x,
               const unsigned short* __restrict__ WpT,
               const unsigned short* __restrict__ WsT,
               float* __restrict__ proj, float* __restrict__ acc, int N) {
    __shared__ unsigned short sWp[128 * 128];   // 32 KB
    __shared__ unsigned short sWs[128 * 128];   // 32 KB

    const int wave = threadIdx.x >> 5;
    const int lane = threadIdx.x & 31;
    const int half = lane >> 4;       // 0: k-offset 0, 1: k-offset +8
    const int r    = lane & 15;
    const int rowBase = (blockIdx.x * 8 + wave) * 16;

    // Wave 0 DMAs both weight matrices into LDS (TDM, TENSORcnt-tracked).
    if (wave == 0) {
        tdm_load_128x128_bf16((unsigned)(unsigned long long)(&sWp[0]), WpT);
        tdm_load_128x128_bf16((unsigned)(unsigned long long)(&sWs[0]), WsT);
        __builtin_amdgcn_s_wait_tensorcnt(0);
    }
    __syncthreads();

    int rowc = rowBase + r;
    if (rowc >= N) rowc = N - 1;                  // compute clamped, store guarded
    const float* xrow = x + (long)rowc * F_TOT;

    // A fragments for the full K=128 (4 steps of 32), kept in registers,
    // reused by both the proj and skip WMMA chains.
    BFrag aF[4];
#pragma unroll
    for (int ks = 0; ks < 4; ++ks) {
        const int k0 = ks * 32 + half * 8;
        float4 f0 = *(const float4*)(xrow + k0);
        float4 f1 = *(const float4*)(xrow + k0 + 4);
        float4 f2 = *(const float4*)(xrow + k0 + 16);
        float4 f3 = *(const float4*)(xrow + k0 + 20);
        aF[ks].s[0]  = f2bf(f0.x); aF[ks].s[1]  = f2bf(f0.y);
        aF[ks].s[2]  = f2bf(f0.z); aF[ks].s[3]  = f2bf(f0.w);
        aF[ks].s[4]  = f2bf(f1.x); aF[ks].s[5]  = f2bf(f1.y);
        aF[ks].s[6]  = f2bf(f1.z); aF[ks].s[7]  = f2bf(f1.w);
        aF[ks].s[8]  = f2bf(f2.x); aF[ks].s[9]  = f2bf(f2.y);
        aF[ks].s[10] = f2bf(f2.z); aF[ks].s[11] = f2bf(f2.w);
        aF[ks].s[12] = f2bf(f3.x); aF[ks].s[13] = f2bf(f3.y);
        aF[ks].s[14] = f2bf(f3.z); aF[ks].s[15] = f2bf(f3.w);
    }

    const bool fullTile = (rowBase + 16 <= N);

#pragma unroll
    for (int nt = 0; nt < 8; ++nt) {
        const int colL = nt * 16 + r;             // this lane's B column
        const unsigned short* wp = sWp + colL * 128;
        const unsigned short* ws = sWs + colL * 128;
        v8f cP = {}, cS = {};
#pragma unroll
        for (int ks = 0; ks < 4; ++ks) {
            const int k0 = ks * 32 + half * 8;
            BFrag bP, bS;
            bP.q[0] = *(const uint4*)(wp + k0);
            bP.q[1] = *(const uint4*)(wp + k0 + 16);
            bS.q[0] = *(const uint4*)(ws + k0);
            bS.q[1] = *(const uint4*)(ws + k0 + 16);
            cP = __builtin_amdgcn_wmma_f32_16x16x32_bf16(
                     false, aF[ks].v, false, bP.v, (short)0, cP, false, false);
            cS = __builtin_amdgcn_wmma_f32_16x16x32_bf16(
                     false, aF[ks].v, false, bS.v, (short)0, cS, false, false);
        }
        const long base = (long)(rowBase + half * 8) * F_TOT + nt * 16 + r;
        if (fullTile) {
#pragma unroll
            for (int j = 0; j < 8; ++j) {
                proj[base + (long)j * F_TOT] = cP[j];
                acc[base + (long)j * F_TOT]  = cS[j];
            }
        } else {
#pragma unroll
            for (int j = 0; j < 8; ++j) {
                if (rowBase + half * 8 + j < N) {
                    proj[base + (long)j * F_TOT] = cP[j];
                    acc[base + (long)j * F_TOT]  = cS[j];
                }
            }
        }
    }
}

// ---------------------------------------------------------------------------
// Kernel 3: per-node attention logits  s_src/s_tgt [N,4]
// ---------------------------------------------------------------------------
__global__ void node_scores(const float* __restrict__ proj,
                            const float* __restrict__ ssrc, const float* __restrict__ stgt,
                            float* __restrict__ s_src, float* __restrict__ s_tgt, int N) {
    int t = blockIdx.x * blockDim.x + threadIdx.x;
    if (t >= N * NH) return;
    int node = t >> 2, h = t & 3;
    const float* p = proj + (long)node * F_TOT + h * 32;
    const float* a = ssrc + h * 32;
    const float* b = stgt + h * 32;
    float as = 0.f, at = 0.f;
#pragma unroll
    for (int f = 0; f < 32; ++f) { float pv = p[f]; as += pv * a[f]; at += pv * b[f]; }
    s_src[t] = as;
    s_tgt[t] = at;
}

// ---------------------------------------------------------------------------
// Kernel 4: edge scores = leaky_relu(s_src[src]+s_tgt[tgt]), global max
// (ordered-uint encoding so atomicMax over unsigned == float max).
// ---------------------------------------------------------------------------
__global__ void edge_scores(const int* __restrict__ ei,
                            const float* __restrict__ s_src, const float* __restrict__ s_tgt,
                            float* __restrict__ scores, unsigned* __restrict__ gmax, int E) {
    int e = blockIdx.x * blockDim.x + threadIdx.x;
    float m = -3.4e38f;
    if (e < E) {
        int s = ei[e], t = ei[E + e];
#pragma unroll
        for (int h = 0; h < NH; ++h) {
            float sc = s_src[s * NH + h] + s_tgt[t * NH + h];
            sc = sc > 0.f ? sc : 0.2f * sc;               // leaky_relu 0.2
            scores[e * NH + h] = sc;
            m = fmaxf(m, sc);
        }
    }
    __shared__ float red[256];
    red[threadIdx.x] = m;
    __syncthreads();
    for (int s2 = 128; s2 > 0; s2 >>= 1) {
        if (threadIdx.x < (unsigned)s2)
            red[threadIdx.x] = fmaxf(red[threadIdx.x], red[threadIdx.x + s2]);
        __syncthreads();
    }
    if (threadIdx.x == 0) {
        unsigned b = __builtin_bit_cast(unsigned, red[0]);
        unsigned o = (b & 0x80000000u) ? ~b : (b | 0x80000000u);
        atomicMax(gmax, o);
    }
}

// ---------------------------------------------------------------------------
// Kernel 5: exp(score - gmax) in place, atomicAdd into denom[tgt,h]
// ---------------------------------------------------------------------------
__global__ void edge_exp_denom(const int* __restrict__ ei, float* __restrict__ scores,
                               float* __restrict__ denom, const unsigned* __restrict__ gmax,
                               int E) {
    int t4 = blockIdx.x * blockDim.x + threadIdx.x;
    if (t4 >= E * NH) return;
    unsigned o  = *gmax;
    unsigned bb = (o & 0x80000000u) ? (o & 0x7FFFFFFFu) : ~o;
    float gm = __builtin_bit_cast(float, bb);
    int e = t4 >> 2, h = t4 & 3;
    int tg = ei[E + e];
    float ev = expf(scores[t4] - gm);
    scores[t4] = ev;                      // now holds exp_scores
    atomicAdd(denom + tg * NH + h, ev);
}

// ---------------------------------------------------------------------------
// Kernel 6: wave-per-edge weighted scatter-add.  Lane l covers features
// 4l..4l+3 (head h = l>>3); float4 gather of proj[src], 4 f32 atomics to acc.
// proj/acc (25.6 MB each) are L2-resident on the 192 MB L2.
// ---------------------------------------------------------------------------
__global__ void __launch_bounds__(256)
aggregate(const int* __restrict__ ei, const float* __restrict__ proj,
          const float* __restrict__ expw, const float* __restrict__ denom,
          float* __restrict__ acc, int E) {
    int gw   = (int)((blockIdx.x * blockDim.x + threadIdx.x) >> 5);   // edge id
    int lane = threadIdx.x & 31;
    if (gw >= E) return;
    int s = ei[gw], t = ei[E + gw];
    int h = lane >> 3;
    float att = expw[gw * NH + h] / (denom[t * NH + h] + 1e-16f);
    float4 p = *(const float4*)(proj + (long)s * F_TOT + lane * 4);
    float* dst = acc + (long)t * F_TOT + lane * 4;
    atomicAdd(dst + 0, p.x * att);
    atomicAdd(dst + 1, p.y * att);
    atomicAdd(dst + 2, p.z * att);
    atomicAdd(dst + 3, p.w * att);
}

// ---------------------------------------------------------------------------
// Kernel 7: out = elu(acc + bias)
// ---------------------------------------------------------------------------
__global__ void finalize(const float* __restrict__ acc, const float* __restrict__ bias,
                         float* __restrict__ out, int total) {
    int t = blockIdx.x * blockDim.x + threadIdx.x;
    if (t >= total) return;
    float v = acc[t] + bias[t & (F_TOT - 1)];
    out[t] = v > 0.f ? v : (expf(v) - 1.f);
}

// ---------------------------------------------------------------------------
extern "C" void kernel_launch(void* const* d_in, const int* in_sizes, int n_in,
                              void* d_out, int out_size, void* d_ws, size_t ws_size,
                              hipStream_t stream) {
    const float* x      = (const float*)d_in[0];
    const int*   ei     = (const int*)d_in[1];      // edge_index (per harness: int)
    const float* Wp     = (const float*)d_in[2];
    const float* ssrc   = (const float*)d_in[3];
    const float* stgt   = (const float*)d_in[4];
    const float* Ws     = (const float*)d_in[5];
    const float* bias   = (const float*)d_in[6];
    float*       out    = (float*)d_out;

    const int N = in_sizes[0] / F_TOT;
    const int E = in_sizes[1] / 2;

    // Workspace carve-up (all chunks are multiples of 256 B).
    char* w = (char*)d_ws;
    float* proj        = (float*)w;  w += (size_t)N * F_TOT * 4;
    float* acc         = (float*)w;  w += (size_t)N * F_TOT * 4;
    unsigned short* WpT = (unsigned short*)w; w += 128 * 128 * 2;
    unsigned short* WsT = (unsigned short*)w; w += 128 * 128 * 2;
    float* s_src       = (float*)w;  w += (size_t)N * NH * 4;
    float* s_tgt       = (float*)w;  w += (size_t)N * NH * 4;
    float* scores      = (float*)w;  w += (size_t)E * NH * 4;
    float* denom       = (float*)w;  w += (size_t)N * NH * 4;
    unsigned* gmax     = (unsigned*)w; w += 256;

    (void)hipMemsetAsync(denom, 0, (size_t)N * NH * 4, stream);
    (void)hipMemsetAsync(gmax, 0, 4, stream);   // 0 < ordered-encoding of any real float

    prep_weights<<<(128 * 128 + 255) / 256, 256, 0, stream>>>(Wp, Ws, WpT, WsT);

    gemm_proj_skip<<<(N + 127) / 128, 256, 0, stream>>>(x, WpT, WsT, proj, acc, N);

    node_scores<<<(N * NH + 255) / 256, 256, 0, stream>>>(proj, ssrc, stgt, s_src, s_tgt, N);

    edge_scores<<<(E + 255) / 256, 256, 0, stream>>>(ei, s_src, s_tgt, scores, gmax, E);

    edge_exp_denom<<<(E * NH + 255) / 256, 256, 0, stream>>>(ei, scores, denom, gmax, E);

    aggregate<<<(E + 7) / 8, 256, 0, stream>>>(ei, proj, scores, denom, acc, E);

    finalize<<<(N * F_TOT + 255) / 256, 256, 0, stream>>>(acc, bias, out, N * F_TOT);
}
